// MultiHeadAttention_70377334112954
// MI455X (gfx1250) — matrix operand order repaired and checked by reading, not compiled, not run on hardware
//
#include <hip/hip_runtime.h>

typedef __attribute__((ext_vector_type(16))) _Float16 v16h;
typedef __attribute__((ext_vector_type(8)))  _Float16 v8h;
typedef __attribute__((ext_vector_type(8)))  float    v8f;

typedef __attribute__((ext_vector_type(4))) unsigned int tdm_g0_t;
typedef __attribute__((ext_vector_type(8))) int          tdm_g1_t;
typedef __attribute__((ext_vector_type(4))) int          tdm_g23_t;

constexpr int Hn = 16;     // heads
constexpr int Dm = 1024;   // model dim
constexpr int DH = 64;     // head dim
constexpr int Bb = 2;      // batch
constexpr int Ss = 2048;   // seq len
constexpr int BS = Bb * Ss;

#if __has_builtin(__builtin_amdgcn_tensor_load_to_lds)
#define USE_TDM 1
#else
#define USE_TDM 0
#endif

__device__ __forceinline__ void wait_tensorcnt0() {
#if __has_builtin(__builtin_amdgcn_s_wait_tensorcnt)
  __builtin_amdgcn_s_wait_tensorcnt(0);
#else
  asm volatile("s_wait_tensorcnt 0x0" ::: "memory");
#endif
}

// ---------------- WMMA fragment helpers (A/B use contiguous k = 16*(lane/16) + j, wave32) ----------------

__device__ __forceinline__ v8f wmma_f16(v16h a, v16h b, v8f c) {
  // (neg_a, A, neg_b, B, c_mod, C, reuse_a, reuse_b)
  return __builtin_amdgcn_wmma_f32_16x16x32_f16(false, a, false, b, (short)0, c,
                                                false, false);
}

// A (16x32, f16 memory, row-major, origin p=(m0,k0), ld in elements)
__device__ __forceinline__ v16h loadA_f16(const _Float16* __restrict__ p, int ld) {
  const int lane = threadIdx.x & 31;
  const int m = lane & 15;
  const int kg = (lane >> 4) * 16;  // 0 or 16
  const _Float16* r = p + (size_t)m * ld;
  v8h lo = *(const v8h*)(r + kg);       // K = kg..kg+7
  v8h hi = *(const v8h*)(r + kg + 8);   // K = kg+8..kg+15
  v16h a;
#pragma unroll
  for (int j = 0; j < 8; ++j) { a[j] = lo[j]; a[8 + j] = hi[j]; }
  return a;
}

// A (16x32, f32 memory, row-major) with on-the-fly f32->f16 convert
__device__ __forceinline__ v16h loadA_f32(const float* __restrict__ p, int ld) {
  const int lane = threadIdx.x & 31;
  const int m = lane & 15;
  const int kg = (lane >> 4) * 16;
  const float* r = p + (size_t)m * ld;
  v8f lo = *(const v8f*)(r + kg);
  v8f hi = *(const v8f*)(r + kg + 8);
  v16h a;
#pragma unroll
  for (int j = 0; j < 8; ++j) {
    a[j]     = (_Float16)lo[j];
    a[8 + j] = (_Float16)hi[j];
  }
  return a;
}

// B (32x16): element (k,n) at p[k*ld + n] (row-major K x N), f32 memory
__device__ __forceinline__ v16h loadB_f32_rowmajor(const float* __restrict__ p, int ld) {
  const int lane = threadIdx.x & 31;
  const int n = lane & 15;
  const int kg = (lane >> 4) * 16;  // 0 or 16
  v16h b;
#pragma unroll
  for (int j = 0; j < 16; ++j) b[j] = (_Float16)p[(size_t)(kg + j) * ld + n];
  return b;
}

// B (32x16): element (k,n) at p[k*ld + n], f16 memory (global fallback path)
__device__ __forceinline__ v16h loadB_f16_rowmajor(const _Float16* __restrict__ p, int ld) {
  const int lane = threadIdx.x & 31;
  const int n = lane & 15;
  const int kg = (lane >> 4) * 16;
  v16h b;
#pragma unroll
  for (int j = 0; j < 16; ++j) b[j] = p[(size_t)(kg + j) * ld + n];
  return b;
}

// B (32x16) from LDS tile (row-major K x N, ld = DH)
__device__ __forceinline__ v16h loadB_lds(const _Float16* p) {
  const int lane = threadIdx.x & 31;
  const int n = lane & 15;
  const int kg = (lane >> 4) * 16;
  v16h b;
#pragma unroll
  for (int j = 0; j < 16; ++j) b[j] = p[(kg + j) * DH + n];
  return b;
}

// B (32x16) from transposed storage: element (k,n) at p[n*ld + k], f16 memory
__device__ __forceinline__ v16h loadB_f16_trans(const _Float16* __restrict__ p, int ld) {
  const int lane = threadIdx.x & 31;
  const int n = lane & 15;
  const int kg = (lane >> 4) * 16;
  return *(const v16h*)(p + (size_t)n * ld + kg);  // 16 contiguous K values
}

// B (32x16) from transposed storage, f32 memory (x @ Wo^T)
__device__ __forceinline__ v16h loadB_f32_trans(const float* __restrict__ p, int ld) {
  const int lane = threadIdx.x & 31;
  const int n = lane & 15;
  const int kg = (lane >> 4) * 16;
  const float* r = p + (size_t)n * ld + kg;
  v8f lo = *(const v8f*)(r);
  v8f hi = *(const v8f*)(r + 8);
  v16h b;
#pragma unroll
  for (int j = 0; j < 8; ++j) { b[j] = (_Float16)lo[j]; b[8 + j] = (_Float16)hi[j]; }
  return b;
}

// C/D 16x16 f32: lane(0..15)=N, VGPR r -> M=r (+8 for lanes 16..31)
__device__ __forceinline__ void storeC_f32(float* __restrict__ p, int ld, v8f c,
                                           float scale) {
  const int lane = threadIdx.x & 31;
  const int n = lane & 15;
  const int mb = (lane >> 4) * 8;
#pragma unroll
  for (int r = 0; r < 8; ++r) p[(size_t)(mb + r) * ld + n] = c[r] * scale;
}

__device__ __forceinline__ void storeC_f16(_Float16* __restrict__ p, int ld, v8f c) {
  const int lane = threadIdx.x & 31;
  const int n = lane & 15;
  const int mb = (lane >> 4) * 8;
#pragma unroll
  for (int r = 0; r < 8; ++r) p[(size_t)(mb + r) * ld + n] = (_Float16)c[r];
}

#if USE_TDM
// Issue one TDM load: 32x64 f16 tile (4KB, rows contiguous) -> LDS.
// D# per CDNA5 ISA ch.8: group0 {count=1, lds_addr, global_addr, type=2},
// group1 {data_size=2B, tensor 64 x 2048 (stride0=64), tile 64 x 32}.
// This toolchain exposes the 6-arg builtin form:
//   (uint32x4 g0, int32x8 g1, int32x4, int32x4, int32x8, i32 cpol)
__device__ __forceinline__ void tdm_load_vtile(const _Float16* gptr,
                                               unsigned lds_byte) {
  const unsigned long long ga = (unsigned long long)(size_t)gptr;
  tdm_g0_t g0;
  g0[0] = 1u;                                   // count=1, user mode
  g0[1] = lds_byte;                             // lds_addr
  g0[2] = (unsigned)(ga & 0xffffffffu);         // global_addr[31:0]
  g0[3] = (unsigned)((ga >> 32) & 0x01ffffffu)  // global_addr[56:32]
          | (2u << 30);                         // type=2 ("image")
  tdm_g1_t g1;
  g1[0] = (int)(1u << 16);          // data_size=1 -> 2 bytes/elem
  g1[1] = (int)((64u & 0xffffu) << 16);   // tensor_dim0 lo16 (=64)
  g1[2] = (int)(2048u << 16);             // tensor_dim0 hi | tensor_dim1 lo16 (=2048)
  g1[3] = (int)(64u << 16);               // tensor_dim1 hi | tile_dim0 (=64)
  g1[4] = 32;                             // tile_dim1=32, tile_dim2=0
  g1[5] = 64;                             // tensor_dim0_stride lo32 (=64 elems)
  g1[6] = 0;                              // stride0 hi | stride1 lo
  g1[7] = 0;                              // stride1 hi
  tdm_g23_t z4 = {0, 0, 0, 0};            // <=2D tensor: groups 2/3 unused
  tdm_g1_t  z8 = {0, 0, 0, 0, 0, 0, 0, 0};
  __builtin_amdgcn_tensor_load_to_lds(g0, g1, z4, z4, z8, 0);
}
#endif

// ---------------- Kernels ----------------

// Y[h] = X @ W[h] : [BS,Dm] x [Dm,DH] -> f16 [H, BS, DH]
// grid (BS/16, H), block 128 (4 waves cover DH=64)
__global__ void __launch_bounds__(128)
proj_kernel(const float* __restrict__ X, const float* __restrict__ W,
            _Float16* __restrict__ Y) {
  const int wave = threadIdx.x >> 5;
  const int m0 = blockIdx.x * 16;
  const int h = blockIdx.y;
  const int n0 = wave * 16;
  const float* Wh = W + (size_t)h * Dm * DH;
  v8f c = {};
  for (int k = 0; k < Dm; k += 32) {
    v16h a = loadA_f32(X + (size_t)m0 * Dm + k, Dm);
    v16h b = loadB_f32_rowmajor(Wh + (size_t)k * DH + n0, DH);
    c = wmma_f16(a, b, c);
  }
  storeC_f16(Y + ((size_t)h * BS + m0) * DH + n0, DH, c);
}

// Fused scores + softmax + attn@V for one (h,b) and a 16-row s-tile.
// grid (Ss/16, H*B), block 128 (4 waves).
__global__ void __launch_bounds__(128)
attn_fused_kernel(const _Float16* __restrict__ q, const _Float16* __restrict__ kk,
                  const _Float16* __restrict__ v, float* __restrict__ attn,
                  _Float16* __restrict__ concat) {
  __shared__ float s_red[4][16];
  __shared__ float g_max[16];
  __shared__ float g_inv[16];
#if USE_TDM
  __shared__ __align__(128) _Float16 vtile[2][32 * DH];  // 2 x 4KB double buffer
#endif

  const int lane = threadIdx.x & 31;
  const int wave = threadIdx.x >> 5;
  const int s0 = blockIdx.x * 16;
  const int hb = blockIdx.y;            // h*B + b
  const int h = hb >> 1;                // B = 2
  const int b = hb & 1;
  const size_t rowbase = (size_t)hb * Ss;
  const int n = lane & 15;
  const int mb = (lane >> 4) * 8;

  const _Float16* qp = q + (rowbase + s0) * DH;
  const _Float16* vbase = v + rowbase * DH;
  float* arow = attn + (rowbase + s0) * Ss;   // this block's [16, Ss] attn rows

  // prefetch the V slab this block will consume in pass 4 (global_prefetch_b8)
  __builtin_prefetch(vbase + (size_t)threadIdx.x * 1024, 0, 1);

  // Q fragments are loop-invariant: load once
  v16h aq0 = loadA_f16(qp, DH);
  v16h aq1 = loadA_f16(qp + 32, DH);

  // ---- pass 1: raw scaled scores -> attn (stays L2-hot), track row max ----
  float rm[8];
#pragma unroll
  for (int r = 0; r < 8; ++r) rm[r] = -3.0e38f;
  for (int t0 = wave * 16; t0 < Ss; t0 += 64) {
    const _Float16* kp = kk + (rowbase + t0) * DH;
    v8f c = {};
    c = wmma_f16(aq0, loadB_f16_trans(kp, DH), c);
    c = wmma_f16(aq1, loadB_f16_trans(kp + 32, DH), c);
    float* op = arow + t0;
#pragma unroll
    for (int r = 0; r < 8; ++r) {
      float sc = c[r] * 0.125f;          // / sqrt(DH)
      rm[r] = fmaxf(rm[r], sc);
      op[(size_t)(mb + r) * Ss + n] = sc;
    }
  }
#pragma unroll
  for (int off = 1; off < 16; off <<= 1) {
#pragma unroll
    for (int r = 0; r < 8; ++r) rm[r] = fmaxf(rm[r], __shfl_xor(rm[r], off, 32));
  }
  if (n == 0) {
#pragma unroll
    for (int r = 0; r < 8; ++r) s_red[wave][mb + r] = rm[r];
  }
  __syncthreads();
  if (threadIdx.x < 16) {
    float m = s_red[0][threadIdx.x];
    for (int w = 1; w < 4; ++w) m = fmaxf(m, s_red[w][threadIdx.x]);
    g_max[threadIdx.x] = m;
  }
  __syncthreads();

  // ---- pass 2: per-row sum of exp (re-reads own just-written rows, L2-hot) ----
  float M[8];
#pragma unroll
  for (int r = 0; r < 8; ++r) M[r] = g_max[mb + r];
  float rs[8] = {};
  for (int t0 = wave * 16; t0 < Ss; t0 += 64) {
    const float* op = arow + t0;
#pragma unroll
    for (int r = 0; r < 8; ++r) rs[r] += __expf(op[(size_t)(mb + r) * Ss + n] - M[r]);
  }
#pragma unroll
  for (int off = 1; off < 16; off <<= 1) {
#pragma unroll
    for (int r = 0; r < 8; ++r) rs[r] += __shfl_xor(rs[r], off, 32);
  }
  __syncthreads();
  if (n == 0) {
#pragma unroll
    for (int r = 0; r < 8; ++r) s_red[wave][mb + r] = rs[r];
  }
  __syncthreads();
  if (threadIdx.x < 16) {
    float s = 0.0f;
    for (int w = 0; w < 4; ++w) s += s_red[w][threadIdx.x];
    g_inv[threadIdx.x] = 1.0f / s;
  }
  __syncthreads();

  // ---- pass 3: write normalized attn (the only HBM-bound attn write) ----
  float inv[8];
#pragma unroll
  for (int r = 0; r < 8; ++r) inv[r] = g_inv[mb + r];
  for (int t0 = wave * 16; t0 < Ss; t0 += 64) {
    float* op = arow + t0;
#pragma unroll
    for (int r = 0; r < 8; ++r) {
      const size_t idx = (size_t)(mb + r) * Ss + n;
      op[idx] = __expf(op[idx] - M[r]) * inv[r];
    }
  }
  __threadfence();   // make all waves' attn rows visible across the block
  __syncthreads();

  // ---- pass 4: weighted = attn[16,Ss] @ V[Ss,DH]; wave w owns 16 of 64 cols ----
  const int n0 = wave * 16;
  v8f acc = {};
#if USE_TDM
  // Double-buffered TDM staging of V tiles (32x64 f16 = 4KB each); the DMA of
  // tile i+1 overlaps WMMA on tile i. One descriptor per tile issued by wave 0;
  // TENSORcnt + block barrier publishes it to all waves.
  const unsigned lds0 = (unsigned)(size_t)(void*)&vtile[0][0];
  constexpr int NT = Ss / 32;   // 64 tiles
  if (wave == 0) tdm_load_vtile(vbase, lds0);
  for (int i = 0; i < NT; ++i) {
    const int cur = i & 1;
    if (wave == 0) wait_tensorcnt0();   // tile i resident in LDS
    __syncthreads();                    // publish tile i; buf cur^1 free
    if (wave == 0 && i + 1 < NT)
      tdm_load_vtile(vbase + (size_t)(i + 1) * 32 * DH, lds0 + (cur ^ 1) * 4096u);
    v16h a  = loadA_f32(arow + i * 32, Ss);           // f32 attn -> f16 frag
    v16h bf = loadB_lds(&vtile[cur][n0]);
    acc = wmma_f16(a, bf, acc);
  }
#else
  const _Float16* vp = vbase + n0;
  for (int t = 0; t < Ss; t += 32) {
    v16h a  = loadA_f32(arow + t, Ss);
    v16h bf = loadB_f16_rowmajor(vp + (size_t)t * DH, DH);
    acc = wmma_f16(a, bf, acc);
  }
#endif
  storeC_f16(concat + ((size_t)(b * Ss + s0)) * Dm + h * DH + n0, Dm, acc);
}

// out = concat @ Wo^T : grid (BS/16, Dm/64)
__global__ void __launch_bounds__(128)
outproj_kernel(const _Float16* __restrict__ concat, const float* __restrict__ Wo,
               float* __restrict__ out) {
  const int wave = threadIdx.x >> 5;
  const int m0 = blockIdx.x * 16;
  const int n0 = (blockIdx.y * 4 + wave) * 16;
  v8f c = {};
  for (int k = 0; k < Dm; k += 32) {
    v16h a = loadA_f16(concat + (size_t)m0 * Dm + k, Dm);
    v16h b = loadB_f32_trans(Wo + (size_t)n0 * Dm + k, Dm);  // B[k,n] = Wo[n,k]
    c = wmma_f16(a, b, c);
  }
  storeC_f32(out + (size_t)m0 * Dm + n0, Dm, c, 1.0f);
}

// ---------------- Launch ----------------

extern "C" void kernel_launch(void* const* d_in, const int* in_sizes, int n_in,
                              void* d_out, int out_size, void* d_ws, size_t ws_size,
                              hipStream_t stream) {
  const float* query = (const float*)d_in[0];
  const float* key_  = (const float*)d_in[1];
  const float* value = (const float*)d_in[2];
  const float* Wq    = (const float*)d_in[3];
  const float* Wk    = (const float*)d_in[4];
  const float* Wv    = (const float*)d_in[5];
  const float* Wo    = (const float*)d_in[6];

  float* out  = (float*)d_out;                       // [B,S,D]
  float* attn = out + (size_t)BS * Dm;               // [H,B,S,S]

  _Float16* qh = (_Float16*)d_ws;                    // [H, BS, DH] f16
  _Float16* kh = qh + (size_t)Hn * BS * DH;
  _Float16* vh = kh + (size_t)Hn * BS * DH;
  _Float16* concat = vh + (size_t)Hn * BS * DH;      // [BS, Dm] f16

  dim3 blk(128);
  proj_kernel<<<dim3(BS / 16, Hn), blk, 0, stream>>>(query, Wq, qh);
  proj_kernel<<<dim3(BS / 16, Hn), blk, 0, stream>>>(key_,  Wk, kh);
  proj_kernel<<<dim3(BS / 16, Hn), blk, 0, stream>>>(value, Wv, vh);
  attn_fused_kernel<<<dim3(Ss / 16, Hn * Bb), blk, 0, stream>>>(qh, kh, vh, attn,
                                                                concat);
  outproj_kernel<<<dim3(BS / 16, Dm / 64), blk, 0, stream>>>(concat, Wo, out);
}
